// Head_47493748359287
// MI455X (gfx1250) — matrix operand order repaired and checked by reading, not compiled, not run on hardware
//
#include <hip/hip_runtime.h>
#include <hip/hip_bf16.h>

// Causal single-head attention for MI455X (gfx1250, wave32, WMMA).
// B=512, T=256, E=384, H=64.  All matmuls via v_wmma_f32_16x16x32_f16.

typedef __attribute__((ext_vector_type(16))) _Float16 v16h;
typedef __attribute__((ext_vector_type(2)))  _Float16 v2h;
typedef __attribute__((ext_vector_type(8)))  float    v8f;

#define BATCH 512
#define TLEN  256
#define EMB   384
#define HDIM  64
#define WFRAG_HALVES (144 * 512)   // 144 fragments x 32 lanes x 16 halves = 73728
// scores are computed in base-2 domain: Q pre-scaled by 1/sqrt(64) * log2(e)
#define QSCALE 0.18033688011112042f

__device__ __forceinline__ v8f wmma16(v16h a, v16h b, v8f c) {
  return __builtin_amdgcn_wmma_f32_16x16x32_f16(
      /*neg_a=*/false, a, /*neg_b=*/false, b,
      /*c_mod=*/(short)0, c, /*reuse_a=*/false, /*reuse_b=*/false);
}

__device__ __forceinline__ float fast_exp2(float v) {
  return __builtin_amdgcn_exp2f(v);     // v_exp_f32 (natively base-2)
}

// A-frag (16x32 f16) K offset for half-pair v (v=0..7 -> elements 2v,2v+1):
// K = 16*hi + 2*v  (pair covers K, K+1)
__device__ __forceinline__ int a_koff(int v, int hi) {
  return (hi << 4) + (v << 1);
}

// ---------------------------------------------------------------------------
// Kernel 0: pack Wk/Wq/Wv (fp32 [384][64]) into f16 B-fragments, kc-major so
// the GEMM kernel's DS offsets within one k-chunk fit the 16-bit immediate.
// Layout: wsW[((kc*12 + m*4 + nt)*32 + lane)*16 + e], B-frag element mapping:
//   N = 16*nt + (lane&15),  K = 32*kc + 16*(lane>>4) + e
// ---------------------------------------------------------------------------
__global__ __launch_bounds__(256) void packw_kernel(
    const float* __restrict__ Wk, const float* __restrict__ Wq,
    const float* __restrict__ Wv, _Float16* __restrict__ wsW) {
  int idx = blockIdx.x * 256 + threadIdx.x;        // 73728 total
  int e    = idx & 15;
  int lane = (idx >> 4) & 31;
  int tid  = idx >> 9;                              // 0..143
  int kc   = tid / 12;
  int rem  = tid % 12;
  int m    = rem >> 2;
  int nt   = rem & 3;
  int N = nt * 16 + (lane & 15);
  int K = kc * 32 + ((lane >> 4) << 4) + e;
  const float* W = (m == 0) ? Wk : ((m == 1) ? Wq : Wv);
  wsW[idx] = (_Float16)W[K * HDIM + N];
}

// ---------------------------------------------------------------------------
// Kernel 1: QKV projection.  One wave per 16-row tile of x; each wave keeps
// all 12 output tiles (K,Q,V x 4 h-tiles) in accumulators so x is read from
// HBM exactly once (201 MB).  W fragments are staged once per block into LDS
// and fed through a depth-2 pipeline of ds_load_b128s so LDS latency hides
// under WMMA execution.  x loads for chunk kc+1 prefetch during chunk kc.
// Writes K (row-major), Q (row-major, pre-scaled by QSCALE), V transposed.
// ---------------------------------------------------------------------------
__global__ __launch_bounds__(256) void qkv_kernel(
    const float* __restrict__ x, const _Float16* __restrict__ wsW,
    _Float16* __restrict__ Kq, _Float16* __restrict__ Qq,
    _Float16* __restrict__ Vt) {
  __shared__ _Float16 sW[WFRAG_HALVES + 1024];      // +pad for depth-2 lookahead

  // cooperative staging: 9216 x 16B
  {
    const uint4* src = (const uint4*)wsW;
    uint4* dst = (uint4*)sW;
    for (int i = threadIdx.x; i < WFRAG_HALVES / 8; i += 256) dst[i] = src[i];
  }
  __syncthreads();

  int lane  = threadIdx.x & 31;
  int wave  = blockIdx.x * 8 + (threadIdx.x >> 5);  // 0..8191 row tiles
  int rbase = wave * 16;
  int hi = lane >> 4, ln = lane & 15;

  v8f acc[12];
#pragma unroll
  for (int t = 0; t < 12; ++t) acc[t] = (v8f){0, 0, 0, 0, 0, 0, 0, 0};

  const float* xr = x + (size_t)(rbase + ln) * EMB;   // A-frag row for this lane

  // prime the A-pipeline with chunk 0
  float2 araw[8];
#pragma unroll
  for (int v = 0; v < 8; ++v)
    araw[v] = *(const float2*)(xr + a_koff(v, hi));

  // prime the B-pipeline (depth 2)
  const _Float16* bp = sW + lane * 16;                // advances 12 frags per kc
  v16h bq0 = *(const v16h*)(bp);
  v16h bq1 = *(const v16h*)(bp + 512);

  for (int kc = 0; kc < 12; ++kc) {
    v16h a;
#pragma unroll
    for (int v = 0; v < 8; ++v) {
      a[2 * v]     = (_Float16)araw[v].x;
      a[2 * v + 1] = (_Float16)araw[v].y;
    }
    if (kc < 11) {                                   // prefetch next x chunk
#pragma unroll
      for (int v = 0; v < 8; ++v)
        araw[v] = *(const float2*)(xr + (kc + 1) * 32 + a_koff(v, hi));
    }
#pragma unroll
    for (int t = 0; t < 12; ++t) {   // t>>2: 0=K,1=Q,2=V ; t&3: h-tile
      v16h bnxt = *(const v16h*)(bp + (t + 2) * 512);  // lookahead (padded)
      acc[t] = wmma16(a, bq0, acc[t]);
      bq0 = bq1;
      bq1 = bnxt;
    }
    bp += 12 * 512;
  }

  // C layout: row = v + 8*hi, col = lane&15
#pragma unroll
  for (int t = 0; t < 12; ++t) {
    int mm = t >> 2, nt = t & 3;
#pragma unroll
    for (int v = 0; v < 8; ++v) {
      int row = rbase + v + 8 * hi;          // global token index
      int col = nt * 16 + ln;                // h index
      if (mm == 0) {
        Kq[(size_t)row * HDIM + col] = (_Float16)acc[t][v];
      } else if (mm == 1) {
        Qq[(size_t)row * HDIM + col] = (_Float16)(acc[t][v] * QSCALE);
      } else {
        int bb = row >> 8, s = row & 255;
        Vt[((size_t)bb * HDIM + col) * TLEN + s] = (_Float16)acc[t][v];
      }
    }
  }
}

// ---------------------------------------------------------------------------
// Kernel 2: flash attention.  One wave per (batch, 16-query tile).
// Per 32-key step: 4 WMMAs (QK^T, K-dim 64) + base-2 online softmax + 4 WMMAs
// (PV).  K and V fragment loads issue together at the top of the step so V
// latency hides under the softmax.  P moves from C-layout to A-frag layout
// via a 1KB/wave LDS buffer.  Key addresses are clamped in-bounds; the causal
// mask zeroes their weights.  Q/K/V (50 MB f16) are L2-resident.
// ---------------------------------------------------------------------------
__global__ __launch_bounds__(256) void attn_kernel(
    const _Float16* __restrict__ Qq, const _Float16* __restrict__ Kq,
    const _Float16* __restrict__ Vt, float* __restrict__ out) {
  __shared__ _Float16 lds[8][16 * 32];     // per-wave P staging (16 rows x 32 keys)
  int lane  = threadIdx.x & 31;
  int wslot = threadIdx.x >> 5;
  int gw    = blockIdx.x * 8 + wslot;      // 0..8191
  int b     = gw >> 4;
  int qbase = (gw & 15) * 16;
  int hi = lane >> 4, ln = lane & 15;

  // Q A-fragments (two K-halves of H=64); Q already carries QSCALE
  v16h qa0, qa1;
  const _Float16* qrow = Qq + ((size_t)b * TLEN + qbase + ln) * HDIM;
#pragma unroll
  for (int v = 0; v < 8; ++v) {
    int k2 = a_koff(v, hi);
    v2h d0 = *(const v2h*)(qrow + k2);
    v2h d1 = *(const v2h*)(qrow + 32 + k2);
    qa0[2 * v] = d0[0]; qa0[2 * v + 1] = d0[1];
    qa1[2 * v] = d1[0]; qa1[2 * v + 1] = d1[1];
  }

  float mrun[8], lrun[8];
  v8f o[4];
#pragma unroll
  for (int v = 0; v < 8; ++v) { mrun[v] = -1e30f; lrun[v] = 0.0f; }
#pragma unroll
  for (int t = 0; t < 4; ++t) o[t] = (v8f){0, 0, 0, 0, 0, 0, 0, 0};

  int nsteps = (qbase + 47) >> 5;          // cover keys 0..qbase+15
  for (int it = 0; it < nsteps; ++it) {
    int sb = it * 32;
    // ---- K and V fragment loads issued together (V covers the softmax) ----
    int s0r = min(sb + ln, TLEN - 1);
    int s1r = min(sb + 16 + ln, TLEN - 1);
    const _Float16* k0p = Kq + ((size_t)b * TLEN + s0r) * HDIM + 16 * hi;
    const _Float16* k1p = Kq + ((size_t)b * TLEN + s1r) * HDIM + 16 * hi;
    v16h kb00 = *(const v16h*)(k0p);
    v16h kb01 = *(const v16h*)(k0p + 32);
    v16h kb10 = *(const v16h*)(k1p);
    v16h kb11 = *(const v16h*)(k1p + 32);
    int soff = min(sb + 16 * hi, TLEN - 16);  // clamped; masked P entries are 0
    v16h vb[4];
#pragma unroll
    for (int t = 0; t < 4; ++t)
      vb[t] = *(const v16h*)(Vt + ((size_t)b * HDIM + t * 16 + ln) * TLEN + soff);

    // ---- scores (base-2 domain): S[16q x 32s] = Q(16x64) @ K^T(64x32) ----
    v8f s0 = (v8f){0, 0, 0, 0, 0, 0, 0, 0};
    v8f s1 = (v8f){0, 0, 0, 0, 0, 0, 0, 0};
    s0 = wmma16(qa0, kb00, s0);
    s0 = wmma16(qa1, kb01, s0);
    s1 = wmma16(qa0, kb10, s1);
    s1 = wmma16(qa1, kb11, s1);

    // ---- causal mask + row max ----
    float cmax[8];
#pragma unroll
    for (int v = 0; v < 8; ++v) {
      int qg = qbase + v + 8 * hi;
      float a0 = s0[v];
      float a1 = s1[v];
      if (sb + ln > qg)      a0 = -1e30f;
      if (sb + 16 + ln > qg) a1 = -1e30f;
      s0[v] = a0; s1[v] = a1;
      cmax[v] = fmaxf(a0, a1);
    }
#pragma unroll
    for (int msk = 1; msk < 16; msk <<= 1)
#pragma unroll
      for (int v = 0; v < 8; ++v)
        cmax[v] = fmaxf(cmax[v], __shfl_xor(cmax[v], msk, 32));

    // ---- online softmax (exp2; v_exp_f32 is natively base-2) ----
    float alpha[8], rs[8];
#pragma unroll
    for (int v = 0; v < 8; ++v) {
      float mn = fmaxf(mrun[v], cmax[v]);
      alpha[v] = fast_exp2(mrun[v] - mn);
      mrun[v]  = mn;
      float p0 = fast_exp2(s0[v] - mn);
      float p1 = fast_exp2(s1[v] - mn);
      s0[v] = p0; s1[v] = p1;
      rs[v] = p0 + p1;
    }
#pragma unroll
    for (int msk = 1; msk < 16; msk <<= 1)
#pragma unroll
      for (int v = 0; v < 8; ++v)
        rs[v] += __shfl_xor(rs[v], msk, 32);
#pragma unroll
    for (int v = 0; v < 8; ++v) lrun[v] = lrun[v] * alpha[v] + rs[v];
#pragma unroll
    for (int t = 0; t < 4; ++t)
#pragma unroll
      for (int v = 0; v < 8; ++v) o[t][v] *= alpha[v];

    // ---- P: C-layout -> A-frag layout via LDS (same-wave, DS in-order) ----
    _Float16* pl = lds[wslot];
#pragma unroll
    for (int v = 0; v < 8; ++v) {
      int row = v + 8 * hi;
      pl[row * 32 + ln]      = (_Float16)s0[v];
      pl[row * 32 + 16 + ln] = (_Float16)s1[v];
    }
    v16h pa;
#pragma unroll
    for (int v = 0; v < 8; ++v) {
      v2h d = *(const v2h*)(pl + ln * 32 + a_koff(v, hi));
      pa[2 * v] = d[0]; pa[2 * v + 1] = d[1];
    }

    // ---- PV: O(16x64) += P(16x32) @ V(32x64) ----
#pragma unroll
    for (int t = 0; t < 4; ++t) o[t] = wmma16(pa, vb[t], o[t]);
  }

  // ---- normalize + store fp32 output ----
  float inv[8];
#pragma unroll
  for (int v = 0; v < 8; ++v) inv[v] = __builtin_amdgcn_rcpf(lrun[v]);
#pragma unroll
  for (int t = 0; t < 4; ++t)
#pragma unroll
    for (int v = 0; v < 8; ++v)
      out[((size_t)b * TLEN + qbase + v + 8 * hi) * HDIM + t * 16 + ln] =
          o[t][v] * inv[v];
}

// ---------------------------------------------------------------------------
extern "C" void kernel_launch(void* const* d_in, const int* in_sizes, int n_in,
                              void* d_out, int out_size, void* d_ws, size_t ws_size,
                              hipStream_t stream) {
  const float* x  = (const float*)d_in[0];
  const float* Wk = (const float*)d_in[1];
  const float* Wq = (const float*)d_in[2];
  const float* Wv = (const float*)d_in[3];
  float* out = (float*)d_out;

  char* ws = (char*)d_ws;
  const size_t QKV_BYTES = (size_t)BATCH * TLEN * HDIM * sizeof(_Float16); // 16 MiB
  _Float16* wsW = (_Float16*)ws;                               // 147456 B
  _Float16* Kq  = (_Float16*)(ws + 147456);
  _Float16* Qq  = (_Float16*)(ws + 147456 + QKV_BYTES);
  _Float16* Vt  = (_Float16*)(ws + 147456 + 2 * QKV_BYTES);

  packw_kernel<<<288, 256, 0, stream>>>(Wk, Wq, Wv, wsW);
  qkv_kernel<<<1024, 256, 0, stream>>>(x, wsW, Kq, Qq, Vt);   // 8192 waves
  attn_kernel<<<1024, 256, 0, stream>>>(Qq, Kq, Vt, out);     // 8192 waves
}
